// QwenImageTransformerBlock_76081050681774
// MI455X (gfx1250) — matrix operand order repaired and checked by reading, not compiled, not run on hardware
//
// QwenImage double-stream transformer block for MI455X (gfx1250, wave32, WMMA).
// bf16 v_wmma_f32_16x16x32_bf16 everywhere (f32 accumulate); weights converted
// f32->bf16 while staging tiles (every weight matrix fits in the 192MB L2);
// A tiles staged via gfx1250 async global->LDS DMA (ASYNCcnt); single-wave
// flash attention with ISA-native fragment layouts loaded straight from global.
#include <hip/hip_runtime.h>
#include <cstdint>
#include <cstddef>

#define DMODEL 3072
#define NHEADS 24
#define HEADD  128
#define SEQ_IMG 2048
#define SEQ_TXT 512
#define SEQ_TOT 2560
#define FFDIM  12288
#define NMOD   18432   // 6*D

typedef __bf16 v16bf __attribute__((ext_vector_type(16)));
typedef float  v8f   __attribute__((ext_vector_type(8)));
typedef unsigned short usv16 __attribute__((ext_vector_type(16)));
typedef unsigned short usv8  __attribute__((ext_vector_type(8)));

__device__ __forceinline__ unsigned short f2bf(float f) {
  unsigned int u = __builtin_bit_cast(unsigned int, f);
  u += 0x7fffu + ((u >> 16) & 1u);   // round-to-nearest-even
  return (unsigned short)(u >> 16);
}

__device__ __forceinline__ unsigned pack2bf(float lo, float hi) {
  return (unsigned)f2bf(lo) | ((unsigned)f2bf(hi) << 16);
}

__device__ __forceinline__ v8f v8f_zero() {
  v8f z;
#pragma unroll
  for (int i = 0; i < 8; ++i) z[i] = 0.0f;
  return z;
}

__device__ __forceinline__ usv16 cat8(usv8 lo, usv8 hi) {
  return __builtin_shufflevector(lo, hi, 0,1,2,3,4,5,6,7,8,9,10,11,12,13,14,15);
}

__device__ __forceinline__ v8f wmma_bf16(usv16 a, usv16 b, v8f c) {
  return __builtin_amdgcn_wmma_f32_16x16x32_bf16(
      false, __builtin_bit_cast(v16bf, a),
      false, __builtin_bit_cast(v16bf, b),
      (short)0, c, false, false);
}

// Async global->LDS copy of 16 bytes per lane (gfx1250, tracked by ASYNCcnt).
__device__ __forceinline__ void async_copy_b128(unsigned lds_addr, const void* gptr) {
  asm volatile("global_load_async_to_lds_b128 %0, %1, off"
               :
               : "v"(lds_addr), "v"(gptr)
               : "memory");
}
__device__ __forceinline__ void wait_async0() {
  asm volatile("s_wait_asynccnt 0" ::: "memory");
}

// ---------------------------------------------------------------- small ops
__global__ __launch_bounds__(256) void silu_kernel(const float* t, float* sil) {
  int i = blockIdx.x * 256 + threadIdx.x;
  if (i < DMODEL) { float x = t[i]; sil[i] = x / (1.0f + __expf(-x)); }
}

// out[j] = bias[j] + sum_k sil[k] * W[k*NMOD + j],  N = 18432
__global__ __launch_bounds__(256) void gemv_mod_kernel(const float* __restrict__ sil,
                                                       const float* __restrict__ W,
                                                       const float* __restrict__ b,
                                                       float* __restrict__ out) {
  int j = blockIdx.x * 256 + threadIdx.x;
  float acc = 0.0f;
  for (int k = 0; k < DMODEL; ++k) acc = fmaf(sil[k], W[(size_t)k * NMOD + j], acc);
  out[j] = acc + b[j];
}

// LayerNorm(row) * (1+scale[c]) + shift[c] -> bf16
__global__ __launch_bounds__(256) void ln_mod_kernel(const float* __restrict__ x,
                                                     const float* __restrict__ shift,
                                                     const float* __restrict__ scale,
                                                     unsigned short* __restrict__ out) {
  const int D = DMODEL;
  int row = blockIdx.x, t = threadIdx.x;
  __shared__ float red[256];
  const float* xr = x + (size_t)row * D;
  float s = 0.0f;
  for (int i = t; i < D; i += 256) s += xr[i];
  red[t] = s; __syncthreads();
  for (int w = 128; w > 0; w >>= 1) { if (t < w) red[t] += red[t + w]; __syncthreads(); }
  float mean = red[0] / D; __syncthreads();
  float vs = 0.0f;
  for (int i = t; i < D; i += 256) { float d = xr[i] - mean; vs += d * d; }
  red[t] = vs; __syncthreads();
  for (int w = 128; w > 0; w >>= 1) { if (t < w) red[t] += red[t + w]; __syncthreads(); }
  float inv = rsqrtf(red[0] / D + 1e-6f);
  for (int i = t; i < D; i += 256) {
    float v = (xr[i] - mean) * inv;
    out[(size_t)row * D + i] = f2bf(v * (1.0f + scale[i]) + shift[i]);
  }
}

// RMSNorm(q,k per head) + RoPE, scatter q,k row-major per head and v transposed.
// grid = (S, H), block = 128 (one thread per head-dim element)
__global__ __launch_bounds__(128) void qkv_post_kernel(const float* __restrict__ qkv,
                                                       const float* __restrict__ cosp,
                                                       const float* __restrict__ sinp,
                                                       const float* __restrict__ wq,
                                                       const float* __restrict__ wk,
                                                       unsigned short* __restrict__ Qb,
                                                       unsigned short* __restrict__ Kb,
                                                       unsigned short* __restrict__ VT,
                                                       int soff) {
  int s = blockIdx.x, h = blockIdx.y, t = threadIdx.x;
  __shared__ float red[128];
  __shared__ float qn[128], kn[128];
  const float* base = qkv + (size_t)s * (3 * DMODEL) + h * HEADD;
  float qv = base[t];
  float kv = base[DMODEL + t];
  float vv = base[2 * DMODEL + t];
  red[t] = qv * qv; __syncthreads();
  for (int w = 64; w > 0; w >>= 1) { if (t < w) red[t] += red[t + w]; __syncthreads(); }
  float rq = rsqrtf(red[0] / HEADD + 1e-6f); __syncthreads();
  red[t] = kv * kv; __syncthreads();
  for (int w = 64; w > 0; w >>= 1) { if (t < w) red[t] += red[t + w]; __syncthreads(); }
  float rk = rsqrtf(red[0] / HEADD + 1e-6f); __syncthreads();
  qn[t] = qv * rq * wq[t];
  kn[t] = kv * rk * wk[t];
  __syncthreads();
  int i = t >> 1;
  float c = cosp[(size_t)s * (HEADD / 2) + i];
  float sn = sinp[(size_t)s * (HEADD / 2) + i];
  float qo, ko;
  if (t & 1) { qo = qn[t] * c + qn[t - 1] * sn; ko = kn[t] * c + kn[t - 1] * sn; }
  else       { qo = qn[t] * c - qn[t + 1] * sn; ko = kn[t] * c - kn[t + 1] * sn; }
  int row = soff + s;
  Qb[((size_t)h * SEQ_TOT + row) * HEADD + t] = f2bf(qo);
  Kb[((size_t)h * SEQ_TOT + row) * HEADD + t] = f2bf(ko);
  VT[((size_t)h * HEADD + t) * SEQ_TOT + row] = f2bf(vv);
}

// ---------------------------------------------------------------- GEMM
// C[M,N] = A(bf16,[M,K]) @ B(f32,[K,N]) + bias, fused epilogue.
// EPI 0: f32 out.  EPI 1: tanh-GELU -> bf16 out.  EPI 2: res + gate[c]*val -> f32.
// block tile 128x64, BK=32, 4 waves (wave tile 64x32).
// A tile: async global->LDS DMA (bf16 already). B tile: f32->bf16 VALU convert,
// stored transposed [n][k] as packed u32 pairs (matches WMMA B fragment layout).
template <int EPI>
__global__ __launch_bounds__(128)
void gemm_bf16_kernel(const unsigned short* __restrict__ A,
                      const float* __restrict__ Bw,
                      const float* __restrict__ bias,
                      void* __restrict__ out,
                      const float* __restrict__ res,
                      const float* __restrict__ gate,
                      int M, int N, int K) {
  __shared__ alignas(32) unsigned short ldsA[128 * 32];  // [m][k] bf16
  __shared__ alignas(32) unsigned short ldsB[64 * 32];   // [n][k] bf16
  const int tid = threadIdx.x;
  const int lane = tid & 31, wid = tid >> 5;
  const int lr = lane & 15, lh = lane >> 4;
  const int wm = (wid >> 1) * 64, wc = (wid & 1) * 32;
  const int n0 = blockIdx.x * 64, m0 = blockIdx.y * 128;

  v8f acc[4][2];
#pragma unroll
  for (int i = 0; i < 4; ++i)
#pragma unroll
    for (int j = 0; j < 2; ++j) acc[i][j] = v8f_zero();

  for (int kb = 0; kb < K; kb += 32) {
    // A tile 128x32 bf16: async DMA, 4x 16B per lane, overlaps with B convert.
#pragma unroll
    for (int j = 0; j < 4; ++j) {
      int c = tid + 128 * j;           // 0..511 16B chunks
      int row = c >> 2;                // 0..127
      int k8 = (c & 3) << 3;           // 0,8,16,24
      const unsigned short* gsrc = A + (size_t)(m0 + row) * K + kb + k8;
      async_copy_b128((unsigned)(size_t)&ldsA[row * 32 + k8], gsrc);
    }

    // B tile (32k x 64n): f32 -> bf16, transposed to [n][k], packed-pair stores.
#pragma unroll
    for (int j = 0; j < 2; ++j) {
      int c = tid + 128 * j;           // 0..255
      int kk = (c >> 4) << 1;          // even k 0..30
      int n4 = (c & 15) << 2;          // 0..60
      const float4 f0 = *(const float4*)(Bw + (size_t)(kb + kk) * N + n0 + n4);
      const float4 f1 = *(const float4*)(Bw + (size_t)(kb + kk + 1) * N + n0 + n4);
      unsigned* dst = (unsigned*)ldsB;
      dst[(n4 + 0) * 16 + (kk >> 1)] = pack2bf(f0.x, f1.x);
      dst[(n4 + 1) * 16 + (kk >> 1)] = pack2bf(f0.y, f1.y);
      dst[(n4 + 2) * 16 + (kk >> 1)] = pack2bf(f0.z, f1.z);
      dst[(n4 + 3) * 16 + (kk >> 1)] = pack2bf(f0.w, f1.w);
    }
    if (kb + 32 < K)  // stream next weight tile toward L2
      __builtin_prefetch(Bw + (size_t)(kb + 32 + (tid >> 4)) * N + n0 + (tid & 15) * 4, 0, 1);

    wait_async0();      // this wave's A DMA done
    __syncthreads();    // all waves' A (async) + B (ds) writes visible

    usv16 afr[4];
#pragma unroll
    for (int mt = 0; mt < 4; ++mt) {
      const unsigned short* ap = &ldsA[(wm + mt * 16 + lr) * 32 + lh * 8];
      afr[mt] = cat8(*(const usv8*)ap, *(const usv8*)(ap + 16));
    }
    usv16 bfr[2];
#pragma unroll
    for (int nt = 0; nt < 2; ++nt) {
      const unsigned short* bp = &ldsB[(wc + nt * 16 + lr) * 32 + lh * 8];
      bfr[nt] = cat8(*(const usv8*)bp, *(const usv8*)(bp + 16));
    }

#pragma unroll
    for (int mt = 0; mt < 4; ++mt)
#pragma unroll
      for (int nt = 0; nt < 2; ++nt)
        acc[mt][nt] = wmma_bf16(afr[mt], bfr[nt], acc[mt][nt]);
    __syncthreads();
  }

#pragma unroll
  for (int mt = 0; mt < 4; ++mt)
#pragma unroll
    for (int nt = 0; nt < 2; ++nt)
#pragma unroll
      for (int v = 0; v < 8; ++v) {
        int row = m0 + wm + mt * 16 + v + lh * 8;
        int col = n0 + wc + nt * 16 + lr;
        float val = acc[mt][nt][v] + bias[col];
        size_t idx = (size_t)row * N + col;
        if (EPI == 0) {
          ((float*)out)[idx] = val;
        } else if (EPI == 1) {
          float u = 0.7978845608028654f * (val + 0.044715f * val * val * val);
          ((unsigned short*)out)[idx] = f2bf(0.5f * val * (1.0f + tanhf(u)));
        } else {
          ((float*)out)[idx] = res[idx] + gate[col] * val;
        }
      }
}

// ---------------------------------------------------------------- attention
// Flash attention, one wave per (16 q-rows, head). K row-major == WMMA B layout
// for Q@K^T; V^T row-major == WMMA B layout for P@V. P transposes via 1KB LDS.
__global__ __launch_bounds__(32)
void attn_kernel(const unsigned short* __restrict__ Qb,
                 const unsigned short* __restrict__ Kb,
                 const unsigned short* __restrict__ VT,
                 unsigned short* __restrict__ O) {
  const int S = SEQ_TOT;
  const float scale = 0.08838834764831845f;  // 1/sqrt(128)
  int qt = blockIdx.x, h = blockIdx.y;
  int lane = threadIdx.x, lr = lane & 15, lh = lane >> 4;
  int q0 = qt * 16;
  __shared__ alignas(32) unsigned short pshare[16 * 32];

  usv16 qf[4];
#pragma unroll
  for (int kk = 0; kk < 4; ++kk) {
    const unsigned short* qp = Qb + ((size_t)h * S + q0 + lr) * HEADD + kk * 32 + lh * 8;
    qf[kk] = cat8(*(const usv8*)qp, *(const usv8*)(qp + 16));
  }

  v8f Oacc[8];
#pragma unroll
  for (int dt = 0; dt < 8; ++dt) Oacc[dt] = v8f_zero();
  float mrow[8], lrow[8];
#pragma unroll
  for (int v = 0; v < 8; ++v) { mrow[v] = -1e30f; lrow[v] = 0.0f; }

  for (int k0 = 0; k0 < S; k0 += 32) {
    v8f l0 = v8f_zero(), l1 = v8f_zero();
#pragma unroll
    for (int kk = 0; kk < 4; ++kk) {
      const unsigned short* kp = Kb + ((size_t)h * S + k0 + lr) * HEADD + kk * 32 + lh * 8;
      l0 = wmma_bf16(qf[kk], cat8(*(const usv8*)kp, *(const usv8*)(kp + 16)), l0);
      l1 = wmma_bf16(qf[kk], cat8(*(const usv8*)(kp + 16 * HEADD), *(const usv8*)(kp + 16 * HEADD + 16)), l1);
    }
    l0 = l0 * scale;
    l1 = l1 * scale;

#pragma unroll
    for (int v = 0; v < 8; ++v) {
      float a0 = l0[v], a1 = l1[v];
      float t = fmaxf(a0, a1);
      t = fmaxf(t, __shfl_xor(t, 1, 32));
      t = fmaxf(t, __shfl_xor(t, 2, 32));
      t = fmaxf(t, __shfl_xor(t, 4, 32));
      t = fmaxf(t, __shfl_xor(t, 8, 32));
      float newm = fmaxf(mrow[v], t);
      float alpha = __expf(mrow[v] - newm);
      float p0 = __expf(a0 - newm);
      float p1 = __expf(a1 - newm);
      float rs = p0 + p1;
      rs += __shfl_xor(rs, 1, 32);
      rs += __shfl_xor(rs, 2, 32);
      rs += __shfl_xor(rs, 4, 32);
      rs += __shfl_xor(rs, 8, 32);
      lrow[v] = lrow[v] * alpha + rs;
      mrow[v] = newm;
      int m = v + lh * 8;
      pshare[m * 32 + lr]      = f2bf(p0);
      pshare[m * 32 + 16 + lr] = f2bf(p1);
#pragma unroll
      for (int dt = 0; dt < 8; ++dt) Oacc[dt][v] *= alpha;
    }
    __syncthreads();

    const unsigned short* pp = &pshare[lr * 32 + lh * 8];
    usv16 pf = cat8(*(const usv8*)pp, *(const usv8*)(pp + 16));
#pragma unroll
    for (int dt = 0; dt < 8; ++dt) {
      const unsigned short* vp = VT + ((size_t)h * HEADD + dt * 16 + lr) * S + k0 + lh * 8;
      Oacc[dt] = wmma_bf16(pf, cat8(*(const usv8*)vp, *(const usv8*)(vp + 16)), Oacc[dt]);
    }
    __syncthreads();
  }

#pragma unroll
  for (int v = 0; v < 8; ++v) {
    float inv = 1.0f / lrow[v];
    unsigned short* orow = O + (size_t)(q0 + v + lh * 8) * DMODEL + h * HEADD;
#pragma unroll
    for (int dt = 0; dt < 8; ++dt) orow[dt * 16 + lr] = f2bf(Oacc[dt][v] * inv);
  }
}

// ---------------------------------------------------------------- host
extern "C" void kernel_launch(void* const* d_in, const int* in_sizes, int n_in,
                              void* d_out, int out_size, void* d_ws, size_t ws_size,
                              hipStream_t stream) {
  (void)in_sizes; (void)n_in; (void)out_size; (void)ws_size;
  const float* hidden_states  = (const float*)d_in[0];
  const float* encoder_hidden = (const float*)d_in[1];
  const float* temb           = (const float*)d_in[2];
  const float* vid_cos        = (const float*)d_in[3];
  const float* vid_sin        = (const float*)d_in[4];
  const float* txt_cos        = (const float*)d_in[5];
  const float* txt_sin        = (const float*)d_in[6];
  const float* img_mod_w      = (const float*)d_in[7];
  const float* img_mod_b      = (const float*)d_in[8];
  const float* txt_mod_w      = (const float*)d_in[9];
  const float* txt_mod_b      = (const float*)d_in[10];
  const float* qkv_w          = (const float*)d_in[11];
  const float* qkv_b          = (const float*)d_in[12];
  const float* add_qkv_w      = (const float*)d_in[13];
  const float* add_qkv_b      = (const float*)d_in[14];
  const float* norm_q_w       = (const float*)d_in[15];
  const float* norm_k_w       = (const float*)d_in[16];
  const float* norm_added_q_w = (const float*)d_in[17];
  const float* norm_added_k_w = (const float*)d_in[18];
  const float* to_out_w       = (const float*)d_in[19];
  const float* to_out_b       = (const float*)d_in[20];
  const float* to_add_out_w   = (const float*)d_in[21];
  const float* to_add_out_b   = (const float*)d_in[22];
  const float* img_mlp_w1     = (const float*)d_in[23];
  const float* img_mlp_b1     = (const float*)d_in[24];
  const float* img_mlp_w2     = (const float*)d_in[25];
  const float* img_mlp_b2     = (const float*)d_in[26];
  const float* txt_mlp_w1     = (const float*)d_in[27];
  const float* txt_mlp_b1     = (const float*)d_in[28];
  const float* txt_mlp_w2     = (const float*)d_in[29];
  const float* txt_mlp_b2     = (const float*)d_in[30];

  char* ws = (char*)d_ws;
  size_t off = 0;
  auto alloc = [&](size_t bytes) -> void* {
    void* p = ws + off;
    off = (off + bytes + 255) & ~(size_t)255;
    return p;
  };
  float* sil      = (float*)alloc((size_t)DMODEL * 4);
  float* img_mod  = (float*)alloc((size_t)NMOD * 4);
  float* txt_mod  = (float*)alloc((size_t)NMOD * 4);
  unsigned short* img_in  = (unsigned short*)alloc((size_t)SEQ_IMG * DMODEL * 2);
  unsigned short* txt_in  = (unsigned short*)alloc((size_t)SEQ_TXT * DMODEL * 2);
  float* qkv_img  = (float*)alloc((size_t)SEQ_IMG * 3 * DMODEL * 4);
  float* qkv_txt  = (float*)alloc((size_t)SEQ_TXT * 3 * DMODEL * 4);
  unsigned short* Qb = (unsigned short*)alloc((size_t)NHEADS * SEQ_TOT * HEADD * 2);
  unsigned short* Kb = (unsigned short*)alloc((size_t)NHEADS * SEQ_TOT * HEADD * 2);
  unsigned short* VT = (unsigned short*)alloc((size_t)NHEADS * HEADD * SEQ_TOT * 2);
  unsigned short* Ob = (unsigned short*)alloc((size_t)SEQ_TOT * DMODEL * 2);
  float* hs_mid   = (float*)alloc((size_t)SEQ_IMG * DMODEL * 4);
  float* es_mid   = (float*)alloc((size_t)SEQ_TXT * DMODEL * 4);
  unsigned short* img_in2 = (unsigned short*)alloc((size_t)SEQ_IMG * DMODEL * 2);
  unsigned short* txt_in2 = (unsigned short*)alloc((size_t)SEQ_TXT * DMODEL * 2);
  unsigned short* img_hid = (unsigned short*)alloc((size_t)SEQ_IMG * FFDIM * 2);
  unsigned short* txt_hid = (unsigned short*)alloc((size_t)SEQ_TXT * FFDIM * 2);

  float* es_out = (float*)d_out;                               // [512, 3072]
  float* hs_out = (float*)d_out + (size_t)SEQ_TXT * DMODEL;    // [2048, 3072]

  // 1) modulation vectors
  silu_kernel<<<dim3((DMODEL + 255) / 256), dim3(256), 0, stream>>>(temb, sil);
  gemv_mod_kernel<<<dim3(NMOD / 256), dim3(256), 0, stream>>>(sil, img_mod_w, img_mod_b, img_mod);
  gemv_mod_kernel<<<dim3(NMOD / 256), dim3(256), 0, stream>>>(sil, txt_mod_w, txt_mod_b, txt_mod);

  // 2) LN + modulate (mod1)
  ln_mod_kernel<<<dim3(SEQ_IMG), dim3(256), 0, stream>>>(hidden_states, img_mod, img_mod + DMODEL, img_in);
  ln_mod_kernel<<<dim3(SEQ_TXT), dim3(256), 0, stream>>>(encoder_hidden, txt_mod, txt_mod + DMODEL, txt_in);

  // 3) QKV projections
  gemm_bf16_kernel<0><<<dim3(3 * DMODEL / 64, SEQ_IMG / 128), dim3(128), 0, stream>>>(
      img_in, qkv_w, qkv_b, qkv_img, nullptr, nullptr, SEQ_IMG, 3 * DMODEL, DMODEL);
  gemm_bf16_kernel<0><<<dim3(3 * DMODEL / 64, SEQ_TXT / 128), dim3(128), 0, stream>>>(
      txt_in, add_qkv_w, add_qkv_b, qkv_txt, nullptr, nullptr, SEQ_TXT, 3 * DMODEL, DMODEL);

  // 4) RMS + RoPE, scatter to per-head layouts (txt rows first)
  qkv_post_kernel<<<dim3(SEQ_TXT, NHEADS), dim3(128), 0, stream>>>(
      qkv_txt, txt_cos, txt_sin, norm_added_q_w, norm_added_k_w, Qb, Kb, VT, 0);
  qkv_post_kernel<<<dim3(SEQ_IMG, NHEADS), dim3(128), 0, stream>>>(
      qkv_img, vid_cos, vid_sin, norm_q_w, norm_k_w, Qb, Kb, VT, SEQ_TXT);

  // 5) flash attention
  attn_kernel<<<dim3(SEQ_TOT / 16, NHEADS), dim3(32), 0, stream>>>(Qb, Kb, VT, Ob);

  // 6) output projections, fused residual + gate1
  gemm_bf16_kernel<2><<<dim3(DMODEL / 64, SEQ_TXT / 128), dim3(128), 0, stream>>>(
      Ob, to_add_out_w, to_add_out_b, es_mid, encoder_hidden, txt_mod + 2 * DMODEL,
      SEQ_TXT, DMODEL, DMODEL);
  gemm_bf16_kernel<2><<<dim3(DMODEL / 64, SEQ_IMG / 128), dim3(128), 0, stream>>>(
      Ob + (size_t)SEQ_TXT * DMODEL, to_out_w, to_out_b, hs_mid, hidden_states,
      img_mod + 2 * DMODEL, SEQ_IMG, DMODEL, DMODEL);

  // 7) LN + modulate (mod2)
  ln_mod_kernel<<<dim3(SEQ_IMG), dim3(256), 0, stream>>>(hs_mid, img_mod + 3 * DMODEL, img_mod + 4 * DMODEL, img_in2);
  ln_mod_kernel<<<dim3(SEQ_TXT), dim3(256), 0, stream>>>(es_mid, txt_mod + 3 * DMODEL, txt_mod + 4 * DMODEL, txt_in2);

  // 8) MLPs: w1+GELU -> bf16, w2 fused residual + gate2 -> d_out
  gemm_bf16_kernel<1><<<dim3(FFDIM / 64, SEQ_IMG / 128), dim3(128), 0, stream>>>(
      img_in2, img_mlp_w1, img_mlp_b1, img_hid, nullptr, nullptr, SEQ_IMG, FFDIM, DMODEL);
  gemm_bf16_kernel<2><<<dim3(DMODEL / 64, SEQ_IMG / 128), dim3(128), 0, stream>>>(
      img_hid, img_mlp_w2, img_mlp_b2, hs_out, hs_mid, img_mod + 5 * DMODEL,
      SEQ_IMG, DMODEL, FFDIM);
  gemm_bf16_kernel<1><<<dim3(FFDIM / 64, SEQ_TXT / 128), dim3(128), 0, stream>>>(
      txt_in2, txt_mlp_w1, txt_mlp_b1, txt_hid, nullptr, nullptr, SEQ_TXT, FFDIM, DMODEL);
  gemm_bf16_kernel<2><<<dim3(DMODEL / 64, SEQ_TXT / 128), dim3(128), 0, stream>>>(
      txt_hid, txt_mlp_w2, txt_mlp_b2, es_out, es_mid, txt_mod + 5 * DMODEL,
      SEQ_TXT, DMODEL, FFDIM);
}
